// DCCF_43344809951755
// MI455X (gfx1250) — matrix operand order repaired and checked by reading, not compiled, not run on hardware
//
#include <hip/hip_runtime.h>
#include <hip/hip_bf16.h>
#include <math.h>

#define D 32
#define KINT 128

typedef __attribute__((ext_vector_type(16))) _Float16 v16h;
typedef __attribute__((ext_vector_type(8)))  _Float16 v8h;
typedef __attribute__((ext_vector_type(8)))  float    v8f;

// ---------------- graph / edge kernels ----------------

__global__ void deg_kernel(const int* __restrict__ h, float* __restrict__ deg, int E) {
    int e = blockIdx.x * blockDim.x + threadIdx.x;
    if (e < E) atomicAdd(&deg[h[e]], 1.0f);
}

__global__ void invsqrt_kernel(float* __restrict__ deg, int n) {
    int i = blockIdx.x * blockDim.x + threadIdx.x;
    if (i < n) { float d = deg[i]; deg[i] = d > 0.0f ? rsqrtf(d) : 0.0f; }
}

__global__ void gvals_kernel(const int* __restrict__ h, const int* __restrict__ t,
                             const float* __restrict__ inv, float* __restrict__ gvals, int E) {
    int e = blockIdx.x * blockDim.x + threadIdx.x;
    if (e < E) gvals[e] = inv[h[e]] * inv[t[e]];
}

// out[h] += val * e_in[t]; val = vals[e] * (rowsum ? 1/rowsum[h] : 1)
// One wave per edge: lanes cover the 32 feature dims -> coalesced 128B gather +
// coalesced 128B global_atomic_add_f32 scatter.
__global__ void spmm_kernel(const int* __restrict__ h, const int* __restrict__ t,
                            const float* __restrict__ vals, const float* __restrict__ rowsum,
                            const float* __restrict__ ein, float* __restrict__ out, int E) {
    long long tid = (long long)blockIdx.x * blockDim.x + threadIdx.x;
    int e = (int)(tid >> 5);
    int d = (int)(tid & 31);
    if (e >= E) return;
    int hh = h[e], tt = t[e];
    float v = vals[e];
    if (rowsum) { float rs = rowsum[hh]; v *= (rs > 0.0f ? 1.0f / rs : 0.0f); }
    atomicAdd(&out[(size_t)hh * D + d], v * ein[(size_t)tt * D + d]);
}

// alpha[e] = (cos(emb[h], emb[t]) + 1)/2 ; rowsum[h] += alpha[e]
__global__ void alpha_kernel(const float* __restrict__ emb, const int* __restrict__ h,
                             const int* __restrict__ t, float* __restrict__ alpha,
                             float* __restrict__ rowsum, int E) {
    int e = blockIdx.x * blockDim.x + threadIdx.x;
    if (e >= E) return;
    const float* a = emb + (size_t)h[e] * D;
    const float* b = emb + (size_t)t[e] * D;
    float na = 0.0f, nb = 0.0f, dp = 0.0f;
#pragma unroll
    for (int k = 0; k < D; ++k) {
        float x = a[k], y = b[k];
        na += x * x; nb += y * y; dp += x * y;
    }
    float inv = rsqrtf(fmaxf(na, 1e-24f)) * rsqrtf(fmaxf(nb, 1e-24f));
    float al = (dp * inv + 1.0f) * 0.5f;
    alpha[e] = al;
    atomicAdd(&rowsum[h[e]], al);
}

// ---------------- elementwise kernels ----------------

__global__ void combine3_kernel(const float* __restrict__ a, const float* __restrict__ b,
                                const float* __restrict__ c, float* __restrict__ out, long long n) {
    long long i = (long long)blockIdx.x * blockDim.x + threadIdx.x;
    if (i < n) out[i] = a[i] + b[i] + c[i];
}

__global__ void accum_kernel(const float* __restrict__ a, float* __restrict__ out, long long n) {
    long long i = (long long)blockIdx.x * blockDim.x + threadIdx.x;
    if (i < n) out[i] += a[i];
}

// ---------------- WMMA intent projection ----------------
// out[16x32 tile] = softmax(e_tile[16x32] @ intent[32x128], axis=1) @ intent^T[128x32]
__global__ void __launch_bounds__(256)
intent_wmma_kernel(const float* __restrict__ e, const float* __restrict__ intent,
                   float* __restrict__ out, int rows) {
    __shared__ _Float16 sIntent[D * KINT];    // row-major [k][n]   (B for GEMM-2), 8 KB
    __shared__ _Float16 sIntentT[KINT * D];   // transposed [n][k]  (B for GEMM-1), 8 KB
    __shared__ _Float16 sP[8][16 * KINT];     // per-wave probs, row-major [m][n], 32 KB

    for (int i = threadIdx.x; i < D * KINT; i += blockDim.x) {
        sIntent[i] = (_Float16)intent[i];               // [k*128 + n]
        int n = i >> 5, k = i & 31;
        sIntentT[i] = (_Float16)intent[k * KINT + n];   // [n*32 + k]
    }
    __syncthreads();

    const int wave = threadIdx.x >> 5;
    const int lane = threadIdx.x & 31;
    const int half = lane >> 4;
    const int l16  = lane & 15;

    int row0 = (blockIdx.x * 8 + wave) * 16;
    if (row0 >= rows) return;   // wave-uniform; EXEC stays all-ones for active waves

    // ---- pack A (16x32 f16): lanes 0-15 hold K {0..7,16..23}, lanes 16-31 K {8..15,24..31}
    const float* rowp = e + (size_t)(row0 + l16) * D;
    float4 fa0 = *(const float4*)(rowp + 8 * half);
    float4 fa1 = *(const float4*)(rowp + 8 * half + 4);
    float4 fa2 = *(const float4*)(rowp + 16 + 8 * half);
    float4 fa3 = *(const float4*)(rowp + 16 + 8 * half + 4);
    v16h A;
    A[0]  = (_Float16)fa0.x; A[1]  = (_Float16)fa0.y; A[2]  = (_Float16)fa0.z; A[3]  = (_Float16)fa0.w;
    A[4]  = (_Float16)fa1.x; A[5]  = (_Float16)fa1.y; A[6]  = (_Float16)fa1.z; A[7]  = (_Float16)fa1.w;
    A[8]  = (_Float16)fa2.x; A[9]  = (_Float16)fa2.y; A[10] = (_Float16)fa2.z; A[11] = (_Float16)fa2.w;
    A[12] = (_Float16)fa3.x; A[13] = (_Float16)fa3.y; A[14] = (_Float16)fa3.z; A[15] = (_Float16)fa3.w;

    // ---- S = A @ intent : 8 tiles of 16x16, one K=32 WMMA each.
    // B pack from transposed copy: per-lane K {8h..8h+7, 16+8h..23+8h} -> two ds_load_b128.
    v8f acc[8];
#pragma unroll
    for (int n = 0; n < 8; ++n) {
        const _Float16* bp = sIntentT + (16 * n + l16) * D + 8 * half;
        v8h blo = *(const v8h*)(bp);
        v8h bhi = *(const v8h*)(bp + 16);
        v16h Bv = __builtin_shufflevector(blo, bhi, 0, 1, 2, 3, 4, 5, 6, 7,
                                          8, 9, 10, 11, 12, 13, 14, 15);
        v8f c = {};
        acc[n] = __builtin_amdgcn_wmma_f32_16x16x32_f16(false, A, false, Bv,
                                                        (short)0, c, false, false);
    }

    // ---- row softmax over the 128 logits (row r+8*half spread over 16-lane half x 8 tiles)
#pragma unroll
    for (int r = 0; r < 8; ++r) {
        float m = -3.0e38f;
#pragma unroll
        for (int n = 0; n < 8; ++n) m = fmaxf(m, acc[n][r]);
#pragma unroll
        for (int off = 1; off <= 8; off <<= 1) m = fmaxf(m, __shfl_xor(m, off, 32));
        float s = 0.0f;
#pragma unroll
        for (int n = 0; n < 8; ++n) { float p = __expf(acc[n][r] - m); acc[n][r] = p; s += p; }
#pragma unroll
        for (int off = 1; off <= 8; off <<= 1) s += __shfl_xor(s, off, 32);
        float is = 1.0f / s;
#pragma unroll
        for (int n = 0; n < 8; ++n) acc[n][r] *= is;
    }

    // ---- stage P (16x128) to LDS row-major f16 (LDS is the cross-lane transpose)
    _Float16* P = sP[wave];
#pragma unroll
    for (int n = 0; n < 8; ++n)
#pragma unroll
        for (int r = 0; r < 8; ++r)
            P[(r + 8 * half) * KINT + 16 * n + l16] = (_Float16)acc[n][r];

    // ---- out = P @ intent^T : K=128 in 4 chunks, N=32 in 2 tiles.
    // A2: per-lane contiguous runs of row-major P. B2: per-lane contiguous runs of sIntent.
    v8f o[2] = {};
#pragma unroll
    for (int c = 0; c < 4; ++c) {
        const _Float16* ap = P + l16 * KINT + 32 * c + 8 * half;
        v8h alo = *(const v8h*)(ap);
        v8h ahi = *(const v8h*)(ap + 16);
        v16h A2 = __builtin_shufflevector(alo, ahi, 0, 1, 2, 3, 4, 5, 6, 7,
                                          8, 9, 10, 11, 12, 13, 14, 15);
#pragma unroll
        for (int t = 0; t < 2; ++t) {
            const _Float16* bp = sIntent + (16 * t + l16) * KINT + 32 * c + 8 * half;
            v8h blo = *(const v8h*)(bp);
            v8h bhi = *(const v8h*)(bp + 16);
            v16h B2 = __builtin_shufflevector(blo, bhi, 0, 1, 2, 3, 4, 5, 6, 7,
                                              8, 9, 10, 11, 12, 13, 14, 15);
            o[t] = __builtin_amdgcn_wmma_f32_16x16x32_f16(false, A2, false, B2,
                                                          (short)0, o[t], false, false);
        }
    }

    // ---- write 16x32 output tile
#pragma unroll
    for (int t = 0; t < 2; ++t)
#pragma unroll
        for (int r = 0; r < 8; ++r)
            out[(size_t)(row0 + r + 8 * half) * D + 16 * t + l16] = o[t][r];
}

// ---------------- loss ----------------

__global__ void loss_kernel(const float* __restrict__ fin, const int* __restrict__ users,
                            const int* __restrict__ pos, const int* __restrict__ neg,
                            float* __restrict__ out, int B, int nUsers) {
    int i = blockIdx.x * blockDim.x + threadIdx.x;
    float v = 0.0f;
    if (i < B) {
        const float* ua = fin + (size_t)users[i] * D;
        const float* pa = fin + (size_t)(nUsers + pos[i]) * D;
        const float* na = fin + (size_t)(nUsers + neg[i]) * D;
        float ps = 0.0f, ns = 0.0f;
#pragma unroll
        for (int d = 0; d < D; ++d) { ps += ua[d] * pa[d]; ns += ua[d] * na[d]; }
        float x = ns - ps;
        v = fmaxf(x, 0.0f) + log1pf(__expf(-fabsf(x)));  // stable softplus
    }
    __shared__ float red[256];
    red[threadIdx.x] = v;
    __syncthreads();
    for (int s = 128; s > 0; s >>= 1) {
        if (threadIdx.x < s) red[threadIdx.x] += red[threadIdx.x + s];
        __syncthreads();
    }
    if (threadIdx.x == 0) atomicAdd(out, red[0] * (1.0f / (float)B));
}

// ---------------- host orchestration ----------------

static inline int nblk(long long n, int tb) { return (int)((n + tb - 1) / tb); }

extern "C" void kernel_launch(void* const* d_in, const int* in_sizes, int n_in,
                              void* d_out, int out_size, void* d_ws, size_t ws_size,
                              hipStream_t stream) {
    const float* user_emb    = (const float*)d_in[0];
    const float* item_emb    = (const float*)d_in[1];
    const float* user_intent = (const float*)d_in[2];
    const float* item_intent = (const float*)d_in[3];
    const int*   all_h       = (const int*)d_in[4];
    const int*   all_t       = (const int*)d_in[5];
    const int*   users       = (const int*)d_in[6];
    const int*   pos_items   = (const int*)d_in[7];
    const int*   neg_items   = (const int*)d_in[8];

    const int nUsers = in_sizes[0] / D;
    const int nItems = in_sizes[1] / D;
    const int nNodes = nUsers + nItems;
    const int E      = in_sizes[4];
    const int B      = in_sizes[6];
    const long long ND = (long long)nNodes * D;

    float* p      = (float*)d_ws;
    float* deg    = p; p += nNodes;
    float* gvals  = p; p += E;
    float* alpha  = p; p += E;
    float* rowsum = p; p += nNodes;
    float* eA     = p; p += ND;
    float* eB     = p; p += ND;
    float* gnn    = p; p += ND;
    float* inte   = p; p += ND;
    float* fin    = p; p += ND;

    const int TB = 256;

    // degree -> inv sqrt -> symmetric norm edge weights
    hipMemsetAsync(deg, 0, (size_t)nNodes * sizeof(float), stream);
    deg_kernel<<<nblk(E, TB), TB, 0, stream>>>(all_h, deg, E);
    invsqrt_kernel<<<nblk(nNodes, TB), TB, 0, stream>>>(deg, nNodes);
    gvals_kernel<<<nblk(E, TB), TB, 0, stream>>>(all_h, all_t, deg, gvals, E);

    // e0 = concat(user_emb, item_emb); final = e0
    hipMemcpyAsync(eA, user_emb, (size_t)nUsers * D * sizeof(float),
                   hipMemcpyDeviceToDevice, stream);
    hipMemcpyAsync(eA + (size_t)nUsers * D, item_emb, (size_t)nItems * D * sizeof(float),
                   hipMemcpyDeviceToDevice, stream);
    hipMemcpyAsync(fin, eA, (size_t)ND * sizeof(float), hipMemcpyDeviceToDevice, stream);

    float* cur = eA;
    float* nxt = eB;
    for (int layer = 0; layer < 2; ++layer) {
        // gnn = norm-adjacency @ cur
        hipMemsetAsync(gnn, 0, (size_t)ND * sizeof(float), stream);
        spmm_kernel<<<nblk((long long)E * D, TB), TB, 0, stream>>>(
            all_h, all_t, gvals, nullptr, cur, gnn, E);

        // intent projections via WMMA (users then items)
        int utiles = (nUsers + 15) / 16, ittiles = (nItems + 15) / 16;
        intent_wmma_kernel<<<(utiles + 7) / 8, 256, 0, stream>>>(
            cur, user_intent, inte, nUsers);
        intent_wmma_kernel<<<(ittiles + 7) / 8, 256, 0, stream>>>(
            cur + (size_t)nUsers * D, item_intent, inte + (size_t)nUsers * D, nItems);

        // nxt = cur + gnn + int_e
        combine3_kernel<<<nblk(ND, TB), TB, 0, stream>>>(cur, gnn, inte, nxt, ND);

        // gaa: adaptive mask from gnn embeddings, then masked SpMM into nxt
        hipMemsetAsync(rowsum, 0, (size_t)nNodes * sizeof(float), stream);
        alpha_kernel<<<nblk(E, TB), TB, 0, stream>>>(gnn, all_h, all_t, alpha, rowsum, E);
        spmm_kernel<<<nblk((long long)E * D, TB), TB, 0, stream>>>(
            all_h, all_t, alpha, rowsum, cur, nxt, E);

        // iaa: adaptive mask from intent embeddings, then masked SpMM into nxt
        hipMemsetAsync(rowsum, 0, (size_t)nNodes * sizeof(float), stream);
        alpha_kernel<<<nblk(E, TB), TB, 0, stream>>>(inte, all_h, all_t, alpha, rowsum, E);
        spmm_kernel<<<nblk((long long)E * D, TB), TB, 0, stream>>>(
            all_h, all_t, alpha, rowsum, cur, nxt, E);

        // final += layer embedding
        accum_kernel<<<nblk(ND, TB), TB, 0, stream>>>(nxt, fin, ND);

        float* tmp = cur; cur = nxt; nxt = tmp;
    }

    // BPR-softplus loss
    hipMemsetAsync(d_out, 0, sizeof(float), stream);
    loss_kernel<<<nblk(B, TB), TB, 0, stream>>>(fin, users, pos_items, neg_items,
                                                (float*)d_out, B, nUsers);
}
